// ModulatedSPADEResnetBlock_55216099557399
// MI455X (gfx1250) — matrix operand order repaired and checked by reading, not compiled, not run on hardware
//
#include <hip/hip_runtime.h>
#include <math.h>

typedef __attribute__((ext_vector_type(16))) _Float16 v16h;
typedef __attribute__((ext_vector_type(8)))  _Float16 h8;
typedef __attribute__((ext_vector_type(8)))  float    v8f;

#define B_   8
#define H_   64
#define W_   64
#define NH_  128
#define SNC_ 35
#define SD_  512

// ---------------------------------------------------------------------------
// Fragment load helpers (wave32; A and B both use contiguous k = 16*half + s)
//   A (16x32 f16): lane 0-15 = row M, halfs {0..7}->K0..7, {8..15}->K8..15
//                  lane 16-31 same rows,   {0..7}->K16..23, {8..15}->K24..31
//   B (32x16 f16): lane 0-15 col N, K=0..15 contiguous; lane 16-31 K=16..31
// ---------------------------------------------------------------------------
__device__ __forceinline__ v16h load_a(const _Float16* p, int a_k0, int a_k1) {
  h8 lo = *(const h8*)(p + a_k0);
  h8 hi = *(const h8*)(p + a_k1);
  v16h av;
  #pragma unroll
  for (int j = 0; j < 8; j++) { av[j] = lo[j]; av[8 + j] = hi[j]; }
  return av;
}
__device__ __forceinline__ v16h load_b(const _Float16* p) {
  h8 lo = *(const h8*)p;
  h8 hi = *(const h8*)(p + 8);
  v16h bv;
  #pragma unroll
  for (int j = 0; j < 8; j++) { bv[j] = lo[j]; bv[8 + j] = hi[j]; }
  return bv;
}
__device__ __forceinline__ v16h zero_a() {
  v16h av;
  #pragma unroll
  for (int j = 0; j < 16; j++) av[j] = (_Float16)0.f;
  return av;
}

// ---------------------------------------------------------------------------
// Instance-norm stats: one block per (b,c), reduce 4096 f32 (NCHW)
// ---------------------------------------------------------------------------
__global__ __launch_bounds__(256) void k_stats(const float* __restrict__ x,
                                               float* __restrict__ mu,
                                               float* __restrict__ rstd) {
  int bc = blockIdx.x;
  const float* p = x + (size_t)bc * 4096;
  float s = 0.f, q = 0.f;
  for (int i = threadIdx.x; i < 4096; i += 256) {
    float v = p[i];
    s += v; q += v * v;
  }
  __shared__ float ss[256], sq[256];
  ss[threadIdx.x] = s; sq[threadIdx.x] = q;
  __syncthreads();
  for (int o = 128; o > 0; o >>= 1) {
    if (threadIdx.x < o) {
      ss[threadIdx.x] += ss[threadIdx.x + o];
      sq[threadIdx.x] += sq[threadIdx.x + o];
    }
    __syncthreads();
  }
  if (threadIdx.x == 0) {
    float m   = ss[0] * (1.f / 4096.f);
    float var = sq[0] * (1.f / 4096.f) - m * m;
    mu[bc]   = m;
    rstd[bc] = rsqrtf(var + 1e-5f);
  }
}

// ---------------------------------------------------------------------------
// Style modulation: s[b,i] = style[b,:] @ (mw[i,:]*ls) + mb[i]
// ---------------------------------------------------------------------------
__global__ void k_style(const float* __restrict__ style,
                        const float* __restrict__ mw,
                        const float* __restrict__ mb,
                        float* __restrict__ s, int I) {
  int idx = blockIdx.x * 256 + threadIdx.x;
  if (idx >= B_ * I) return;
  int b = idx / I, i = idx % I;
  const float ls = 0.04419417382415922f; // 1/sqrt(512)
  float acc = mb[i];
  const float* st = style + (size_t)b * SD_;
  const float* wr = mw + (size_t)i * SD_;
  for (int d = 0; d < SD_; d++) acc += st[d] * wr[d] * ls;
  s[idx] = acc;
}

// ---------------------------------------------------------------------------
// wsum[o,i] = sum_k w[o,i,k]^2   (K2 = 9 or 1)
// ---------------------------------------------------------------------------
__global__ void k_wsum(const float* __restrict__ w, float* __restrict__ wsum,
                       int O, int I, int K2) {
  int idx = blockIdx.x * 256 + threadIdx.x;
  if (idx >= O * I) return;
  const float* p = w + (size_t)idx * K2;
  float a = 0.f;
  for (int t = 0; t < K2; t++) a += p[t] * p[t];
  wsum[idx] = a;
}

// ---------------------------------------------------------------------------
// demod[b,o] = rsqrt( cs^2 * sum_i s[b,i]^2 * wsum[o,i] + 1e-8 )
// ---------------------------------------------------------------------------
__global__ void k_demod(const float* __restrict__ s,
                        const float* __restrict__ wsum,
                        float* __restrict__ demod, int I, float cs2) {
  int idx = blockIdx.x * 256 + threadIdx.x;
  if (idx >= B_ * 256) return;
  int b = idx / 256, o = idx % 256;
  float a = 0.f;
  const float* sp = s + (size_t)b * I;
  const float* wp = wsum + (size_t)o * I;
  for (int i = 0; i < I; i++) { float sv = sp[i]; a += sv * sv * wp[i]; }
  demod[idx] = rsqrtf(a * cs2 + 1e-8f);
}

// ---------------------------------------------------------------------------
// Weight convert: f32 OIHW -> f16 [tap][o][i] (i innermost), times scale
// ---------------------------------------------------------------------------
__global__ void k_wcvt(const float* __restrict__ src, _Float16* __restrict__ dst,
                       int O, int I, int K2, float scale) {
  int idx = blockIdx.x * 256 + threadIdx.x;
  if (idx >= O * I * K2) return;
  int i = idx % I;
  int o = (idx / I) % O;
  int t = idx / (I * O);
  dst[idx] = (_Float16)(src[((size_t)o * I + i) * K2 + t] * scale);
}

// ---------------------------------------------------------------------------
// Seg conv 3x3: 35 -> 128, +bias, ReLU, out NHWC f16. One block per (y,b).
// seg row tile staged in LDS (emits ds ops).
// ---------------------------------------------------------------------------
__global__ __launch_bounds__(256) void k_segconv(const float* __restrict__ seg,
                                                 const float* __restrict__ sw,
                                                 const float* __restrict__ sb,
                                                 _Float16* __restrict__ actv) {
  int y = blockIdx.x, b = blockIdx.y;
  __shared__ float ls[SNC_ * 3 * W_];
  for (int e = threadIdx.x; e < SNC_ * 3 * W_; e += 256) {
    int c  = e / (3 * W_);
    int r  = (e / W_) % 3;
    int xx = e % W_;
    int ys = y + r - 1;
    ls[e] = (ys >= 0 && ys < H_)
              ? seg[(((size_t)b * SNC_ + c) * H_ + ys) * W_ + xx] : 0.f;
  }
  __syncthreads();
  int x  = threadIdx.x & 63;
  int cq = threadIdx.x >> 6;        // 4 channel quarters of 32
  _Float16* op = actv + ((size_t)(b * H_ + y) * W_ + x) * NH_ + cq * 32;
  for (int j = 0; j < 32; j++) {
    int c = cq * 32 + j;
    float a = sb[c];
    const float* wp = sw + (size_t)c * (SNC_ * 9);
    __builtin_prefetch(wp + SNC_ * 9, 0, 3);   // next channel's weights
    for (int i = 0; i < SNC_; i++) {
      #pragma unroll
      for (int ky = 0; ky < 3; ky++) {
        #pragma unroll
        for (int kx = 0; kx < 3; kx++) {
          int xs = x + kx - 1;
          float sv = (xs >= 0 && xs < W_) ? ls[(i * 3 + ky) * W_ + xs] : 0.f;
          a += sv * wp[(i * 3 + ky) * 3 + kx];
        }
      }
    }
    op[j] = (_Float16)(a > 0.f ? a : 0.f);
  }
}

// ---------------------------------------------------------------------------
// SPADE: gamma/beta 3x3 convs (128 -> C) as dual implicit GEMM with WMMA,
// fused instance-norm + affine + optional lrelu + input-modulation scale.
// Block: 256 thr = 8 waves (2 M-groups x 4 N-groups); tile 64 px x 256 oc.
// Grid: (64 rows, C/256, B).
// ---------------------------------------------------------------------------
__global__ __launch_bounds__(256) void k_spade(
    const _Float16* __restrict__ actv,  // [B,H,W,128] NHWC f16
    const _Float16* __restrict__ wg,    // [9,C,128] f16
    const _Float16* __restrict__ wb,    // [9,C,128] f16
    const float* __restrict__ gb, const float* __restrict__ bb,  // [C]
    const float* __restrict__ xsrc,     // [B,C,H,W] NCHW f32
    const float* __restrict__ mu, const float* __restrict__ rstd, // [B*C]
    const float* __restrict__ s,        // [B*C] modulation of NEXT conv
    _Float16* __restrict__ out,         // [B,H,W,C] NHWC f16
    int C, int do_lrelu) {
  int y    = blockIdx.x;
  int ocb  = blockIdx.y * 256;
  int b    = blockIdx.z;
  int tid  = threadIdx.x;
  int lane = tid & 31, wave = tid >> 5;
  int wm = wave & 1, wn = wave >> 1;
  int half = lane >> 4, m16 = lane & 15;
  int px0 = wm * 32;

  v8f zero = {};
  v8f accG[2][4], accB[2][4];
  #pragma unroll
  for (int i = 0; i < 2; i++)
    #pragma unroll
    for (int j = 0; j < 4; j++) { accG[i][j] = zero; accB[i][j] = zero; }

  const int a_k0 = half * 16, a_k1 = half * 16 + 8;  // A frag contiguous run
  const int b_k0 = half * 16;                       // B frag contiguous run

  for (int t = 0; t < 9; t++) {
    int ky = t / 3, kx = t % 3;
    int ysrc = y + ky - 1;
    if (ysrc < 0 || ysrc >= H_) continue;
    const _Float16* arow  = actv + (size_t)(b * H_ + ysrc) * W_ * NH_;
    const _Float16* wgt_g = wg + (size_t)t * C * NH_;
    const _Float16* wgt_b = wb + (size_t)t * C * NH_;
    // prefetch next tap's weight rows for this wave's oc columns
    if (t < 8) {
      __builtin_prefetch(wgt_g + (size_t)C * NH_ +
                         (size_t)(ocb + wn * 64 + m16) * NH_, 0, 3);
      __builtin_prefetch(wgt_b + (size_t)C * NH_ +
                         (size_t)(ocb + wn * 64 + m16) * NH_, 0, 3);
    }
    for (int ic = 0; ic < NH_ / 32; ic++) {
      int c0 = ic * 32;
      v16h a[2];
      if (kx == 1) {
        #pragma unroll
        for (int ms = 0; ms < 2; ms++)
          a[ms] = load_a(arow + (size_t)(px0 + ms * 16 + m16) * NH_ + c0,
                         a_k0, a_k1);
      } else {
        #pragma unroll
        for (int ms = 0; ms < 2; ms++) {
          int xs = px0 + ms * 16 + m16 + kx - 1;
          a[ms] = (xs >= 0 && xs < W_)
                    ? load_a(arow + (size_t)xs * NH_ + c0, a_k0, a_k1)
                    : zero_a();
        }
      }
      #pragma unroll
      for (int ns = 0; ns < 4; ns++) {
        int oc = ocb + wn * 64 + ns * 16 + m16;
        v16h bg = load_b(wgt_g + (size_t)oc * NH_ + c0 + b_k0);
        v16h bv = load_b(wgt_b + (size_t)oc * NH_ + c0 + b_k0);
        #pragma unroll
        for (int ms = 0; ms < 2; ms++) {
          accG[ms][ns] = __builtin_amdgcn_wmma_f32_16x16x32_f16(
              false, a[ms], false, bg, (short)0, accG[ms][ns], false, false);
          accB[ms][ns] = __builtin_amdgcn_wmma_f32_16x16x32_f16(
              false, a[ms], false, bv, (short)0, accB[ms][ns], false, false);
        }
      }
    }
  }

  // epilogue: D frag -> M = vgpr + 8*half (pixel), N = lane&15 (channel)
  #pragma unroll
  for (int ms = 0; ms < 2; ms++) {
    #pragma unroll
    for (int ns = 0; ns < 4; ns++) {
      int c = ocb + wn * 64 + ns * 16 + m16;
      float gbias = gb[c], bbias = bb[c];
      float muv = mu[b * C + c], rs = rstd[b * C + c], sv = s[b * C + c];
      #pragma unroll
      for (int v = 0; v < 8; v++) {
        int px = px0 + ms * 16 + v + 8 * half;
        float gamma = accG[ms][ns][v] + gbias;
        float beta  = accB[ms][ns][v] + bbias;
        float xv = xsrc[(((size_t)b * C + c) * H_ + y) * W_ + px];
        float val = (xv - muv) * rs * (1.f + gamma) + beta;
        if (do_lrelu) val = val > 0.f ? val : 0.2f * val;
        out[((size_t)(b * H_ + y) * W_ + px) * C + c] = (_Float16)(val * sv);
      }
    }
  }
}

// ---------------------------------------------------------------------------
// Modulated conv (3x3 or 1x1) as implicit GEMM with WMMA; demod epilogue,
// optional shortcut add, output NCHW f32. O fixed = 256.
// Grid: (64 rows, 1, B); block 256 thr; tile 64 px x 256 oc.
// ---------------------------------------------------------------------------
__global__ __launch_bounds__(256) void k_modconv(
    const _Float16* __restrict__ xin,   // [B,H,W,I] NHWC f16 (pre-scaled by s)
    const _Float16* __restrict__ wf,    // [taps,256,I] f16 (times conv_scale)
    const float* __restrict__ demod,    // [B*256]
    const float* __restrict__ addsrc,   // nullable, [B,256,H,W] NCHW f32
    float* __restrict__ out,            // [B,256,H,W] NCHW f32
    int I, int taps) {
  int y = blockIdx.x, b = blockIdx.z;
  int tid = threadIdx.x;
  int lane = tid & 31, wave = tid >> 5;
  int wm = wave & 1, wn = wave >> 1;
  int half = lane >> 4, m16 = lane & 15;
  int px0 = wm * 32;

  v8f zero = {};
  v8f acc[2][4];
  #pragma unroll
  for (int i = 0; i < 2; i++)
    #pragma unroll
    for (int j = 0; j < 4; j++) acc[i][j] = zero;

  const int a_k0 = half * 16, a_k1 = half * 16 + 8;
  const int b_k0 = half * 16;
  const int nic = I / 32;

  for (int t = 0; t < taps; t++) {
    int ky = (taps == 9) ? t / 3 : 1;
    int kx = (taps == 9) ? t % 3 : 1;
    int ysrc = y + ky - 1;
    if (ysrc < 0 || ysrc >= H_) continue;
    const _Float16* arow = xin + (size_t)(b * H_ + ysrc) * W_ * I;
    const _Float16* wt   = wf + (size_t)t * 256 * I;
    if (t + 1 < taps) {   // prefetch next tap's weights for our columns
      __builtin_prefetch(wt + (size_t)256 * I +
                         (size_t)(wn * 64 + m16) * I, 0, 3);
    }
    for (int ic = 0; ic < nic; ic++) {
      int c0 = ic * 32;
      v16h a[2];
      if (kx == 1) {
        #pragma unroll
        for (int ms = 0; ms < 2; ms++)
          a[ms] = load_a(arow + (size_t)(px0 + ms * 16 + m16) * I + c0,
                         a_k0, a_k1);
      } else {
        #pragma unroll
        for (int ms = 0; ms < 2; ms++) {
          int xs = px0 + ms * 16 + m16 + kx - 1;
          a[ms] = (xs >= 0 && xs < W_)
                    ? load_a(arow + (size_t)xs * I + c0, a_k0, a_k1)
                    : zero_a();
        }
      }
      #pragma unroll
      for (int ns = 0; ns < 4; ns++) {
        int oc = wn * 64 + ns * 16 + m16;
        v16h bv = load_b(wt + (size_t)oc * I + c0 + b_k0);
        #pragma unroll
        for (int ms = 0; ms < 2; ms++) {
          acc[ms][ns] = __builtin_amdgcn_wmma_f32_16x16x32_f16(
              false, a[ms], false, bv, (short)0, acc[ms][ns], false, false);
        }
      }
    }
  }

  #pragma unroll
  for (int ms = 0; ms < 2; ms++) {
    #pragma unroll
    for (int ns = 0; ns < 4; ns++) {
      int c = wn * 64 + ns * 16 + m16;
      float dm = demod[b * 256 + c];
      #pragma unroll
      for (int v = 0; v < 8; v++) {
        int px = px0 + ms * 16 + v + 8 * half;
        size_t idx = (((size_t)b * 256 + c) * H_ + y) * W_ + px;
        float val = acc[ms][ns][v] * dm;
        if (addsrc) val += addsrc[idx];
        out[idx] = val;
      }
    }
  }
}

// ---------------------------------------------------------------------------
extern "C" void kernel_launch(void* const* d_in, const int* in_sizes, int n_in,
                              void* d_out, int out_size, void* d_ws, size_t ws_size,
                              hipStream_t stream) {
  const float* x     = (const float*)d_in[0];
  const float* seg   = (const float*)d_in[1];
  const float* style = (const float*)d_in[2];
  const float* n0_sw = (const float*)d_in[3];
  const float* n0_sb = (const float*)d_in[4];
  const float* n0_gw = (const float*)d_in[5];
  const float* n0_gb = (const float*)d_in[6];
  const float* n0_bw = (const float*)d_in[7];
  const float* n0_bb = (const float*)d_in[8];
  const float* n1_sw = (const float*)d_in[9];
  const float* n1_sb = (const float*)d_in[10];
  const float* n1_gw = (const float*)d_in[11];
  const float* n1_gb = (const float*)d_in[12];
  const float* n1_bw = (const float*)d_in[13];
  const float* n1_bb = (const float*)d_in[14];
  const float* ns_sw = (const float*)d_in[15];
  const float* ns_sb = (const float*)d_in[16];
  const float* ns_gw = (const float*)d_in[17];
  const float* ns_gb = (const float*)d_in[18];
  const float* ns_bw = (const float*)d_in[19];
  const float* ns_bb = (const float*)d_in[20];
  const float* w0    = (const float*)d_in[21];
  const float* m0w   = (const float*)d_in[22];
  const float* m0b   = (const float*)d_in[23];
  const float* w1    = (const float*)d_in[24];
  const float* m1w   = (const float*)d_in[25];
  const float* m1b   = (const float*)d_in[26];
  const float* wsc   = (const float*)d_in[27];
  const float* msw   = (const float*)d_in[28];
  const float* msb   = (const float*)d_in[29];
  float* out = (float*)d_out;

  // workspace carve-up
  char* base = (char*)d_ws;
  size_t off = 0;
  auto alloc = [&](size_t bytes) -> void* {
    void* p = base + off;
    off = (off + bytes + 255) & ~(size_t)255;
    return p;
  };
  float*     mu_x  = (float*)alloc(4096 * 4);
  float*     rs_x  = (float*)alloc(4096 * 4);
  float*     mu_d  = (float*)alloc(2048 * 4);
  float*     rs_d  = (float*)alloc(2048 * 4);
  float*     s0    = (float*)alloc(8 * 512 * 4);
  float*     s1    = (float*)alloc(8 * 256 * 4);
  float*     ss    = (float*)alloc(8 * 512 * 4);
  float*     wsum0 = (float*)alloc(256 * 512 * 4);
  float*     wsum1 = (float*)alloc(256 * 256 * 4);
  float*     wsums = (float*)alloc(256 * 512 * 4);
  float*     dm0   = (float*)alloc(2048 * 4);
  float*     dm1   = (float*)alloc(2048 * 4);
  float*     dms   = (float*)alloc(2048 * 4);
  _Float16*  w0f   = (_Float16*)alloc((size_t)9 * 256 * 512 * 2);
  _Float16*  w1f   = (_Float16*)alloc((size_t)9 * 256 * 256 * 2);
  _Float16*  wsf   = (_Float16*)alloc((size_t)256 * 512 * 2);
  _Float16*  g0f   = (_Float16*)alloc((size_t)9 * 512 * 128 * 2);
  _Float16*  b0f   = (_Float16*)alloc((size_t)9 * 512 * 128 * 2);
  _Float16*  g1f   = (_Float16*)alloc((size_t)9 * 256 * 128 * 2);
  _Float16*  b1f   = (_Float16*)alloc((size_t)9 * 256 * 128 * 2);
  _Float16*  gsf   = (_Float16*)alloc((size_t)9 * 512 * 128 * 2);
  _Float16*  bsf   = (_Float16*)alloc((size_t)9 * 512 * 128 * 2);
  _Float16*  actv0 = (_Float16*)alloc((size_t)8 * 64 * 64 * 128 * 2);
  _Float16*  actv1 = (_Float16*)alloc((size_t)8 * 64 * 64 * 128 * 2);
  _Float16*  actvs = (_Float16*)alloc((size_t)8 * 64 * 64 * 128 * 2);
  _Float16*  xin0  = (_Float16*)alloc((size_t)8 * 64 * 64 * 512 * 2);
  _Float16*  xin1  = (_Float16*)alloc((size_t)8 * 64 * 64 * 256 * 2);
  _Float16*  xins  = (_Float16*)alloc((size_t)8 * 64 * 64 * 512 * 2);
  float*     xs_buf = (float*)alloc((size_t)8 * 256 * 64 * 64 * 4);
  float*     dx0    = (float*)alloc((size_t)8 * 256 * 64 * 64 * 4);
  (void)ws_size; (void)n_in; (void)in_sizes; (void)out_size;

  const float cs0 = 1.0f / sqrtf(512.f * 9.f);
  const float cs1 = 1.0f / sqrtf(256.f * 9.f);
  const float css = 1.0f / sqrtf(512.f);

  // 1) instance-norm stats of x
  k_stats<<<4096, 256, 0, stream>>>(x, mu_x, rs_x);

  // 2) style linears
  k_style<<<16, 256, 0, stream>>>(style, m0w, m0b, s0, 512);
  k_style<<<8,  256, 0, stream>>>(style, m1w, m1b, s1, 256);
  k_style<<<16, 256, 0, stream>>>(style, msw, msb, ss, 512);

  // 3) per-(o,i) weight energy + demod factors
  k_wsum<<<512, 256, 0, stream>>>(w0,  wsum0, 256, 512, 9);
  k_wsum<<<256, 256, 0, stream>>>(w1,  wsum1, 256, 256, 9);
  k_wsum<<<512, 256, 0, stream>>>(wsc, wsums, 256, 512, 1);
  k_demod<<<8, 256, 0, stream>>>(s0, wsum0, dm0, 512, cs0 * cs0);
  k_demod<<<8, 256, 0, stream>>>(s1, wsum1, dm1, 256, cs1 * cs1);
  k_demod<<<8, 256, 0, stream>>>(ss, wsums, dms, 512, css * css);

  // 4) weight conversions to f16 [tap][o][i]
  k_wcvt<<<(9*256*512+255)/256, 256, 0, stream>>>(w0,  w0f, 256, 512, 9, cs0);
  k_wcvt<<<(9*256*256+255)/256, 256, 0, stream>>>(w1,  w1f, 256, 256, 9, cs1);
  k_wcvt<<<(256*512+255)/256,   256, 0, stream>>>(wsc, wsf, 256, 512, 1, css);
  k_wcvt<<<(9*512*128+255)/256, 256, 0, stream>>>(n0_gw, g0f, 512, 128, 9, 1.f);
  k_wcvt<<<(9*512*128+255)/256, 256, 0, stream>>>(n0_bw, b0f, 512, 128, 9, 1.f);
  k_wcvt<<<(9*256*128+255)/256, 256, 0, stream>>>(n1_gw, g1f, 256, 128, 9, 1.f);
  k_wcvt<<<(9*256*128+255)/256, 256, 0, stream>>>(n1_bw, b1f, 256, 128, 9, 1.f);
  k_wcvt<<<(9*512*128+255)/256, 256, 0, stream>>>(ns_gw, gsf, 512, 128, 9, 1.f);
  k_wcvt<<<(9*512*128+255)/256, 256, 0, stream>>>(ns_bw, bsf, 512, 128, 9, 1.f);

  // 5) seg convs (shared segmentation map, three weight sets)
  k_segconv<<<dim3(64, 8), 256, 0, stream>>>(seg, n0_sw, n0_sb, actv0);
  k_segconv<<<dim3(64, 8), 256, 0, stream>>>(seg, n1_sw, n1_sb, actv1);
  k_segconv<<<dim3(64, 8), 256, 0, stream>>>(seg, ns_sw, ns_sb, actvs);

  // 6) SPADE(ns) (no lrelu) and SPADE(n0) (lrelu); s-scaled f16 NHWC out
  k_spade<<<dim3(64, 2, 8), 256, 0, stream>>>(actvs, gsf, bsf, ns_gb, ns_bb,
                                              x, mu_x, rs_x, ss, xins, 512, 0);
  k_spade<<<dim3(64, 2, 8), 256, 0, stream>>>(actv0, g0f, b0f, n0_gb, n0_bb,
                                              x, mu_x, rs_x, s0, xin0, 512, 1);

  // 7) shortcut 1x1 modconv and main 3x3 modconv0
  k_modconv<<<dim3(64, 1, 8), 256, 0, stream>>>(xins, wsf, dms, nullptr,
                                                xs_buf, 512, 1);
  k_modconv<<<dim3(64, 1, 8), 256, 0, stream>>>(xin0, w0f, dm0, nullptr,
                                                dx0, 512, 9);

  // 8) stats of dx, SPADE(n1), modconv1 (+ shortcut add) -> d_out (NCHW f32)
  k_stats<<<2048, 256, 0, stream>>>(dx0, mu_d, rs_d);
  k_spade<<<dim3(64, 1, 8), 256, 0, stream>>>(actv1, g1f, b1f, n1_gb, n1_bb,
                                              dx0, mu_d, rs_d, s1, xin1, 256, 1);
  k_modconv<<<dim3(64, 1, 8), 256, 0, stream>>>(xin1, w1f, dm1, xs_buf,
                                                out, 256, 9);
}
